// BitLinear_7499012899209
// MI455X (gfx1250) — matrix operand order repaired and checked by reading, not compiled, not run on hardware
//
#include <hip/hip_runtime.h>
#include <hip/hip_bf16.h>

// ---------------------------------------------------------------------------
// BitLinear forward for MI455X (gfx1250, wave32, WMMA + Tensor Data Mover).
//   out[m,n] = sum_k x_bf16[m,k] * wq_bf16[n,k] + bias[n]
// M=8192, N=16384, K=4096, ternary group=128.
// Block tile 256x128 (8 waves, 64x64 each -> 16 WMMAs/wave/K-step).
// Tiles staged by TDM (TENSORcnt), double-buffered LDS overlaps DMA w/ math.
// ---------------------------------------------------------------------------

typedef __attribute__((ext_vector_type(16))) __bf16   bf16x16;
typedef __attribute__((ext_vector_type(8)))  float    f32x8;
typedef __attribute__((ext_vector_type(4)))  unsigned u32x4;
typedef __attribute__((ext_vector_type(8)))  int      i32x8;
typedef __attribute__((ext_vector_type(4)))  int      i32x4;

#define GEMM_M 8192
#define GEMM_N 16384
#define GEMM_K 4096
#define KSTEP  32
#define LDSK   40    // halves per LDS row: 32 data + 8 pad (80B) == TDM 64B+16B pad
#define TILE_M 256
#define TILE_N 128

union FragU { uint4 u[2]; bf16x16 v; };

__device__ __forceinline__ unsigned short f2bf(float f) {
    unsigned u = __builtin_bit_cast(unsigned, f);
    unsigned r = u + 0x7fffu + ((u >> 16) & 1u);   // round-to-nearest-even
    return (unsigned short)(r >> 16);
}

// -------------------- kernel 1: groupwise ternary quant of W ---------------
__global__ __launch_bounds__(256) void bitlinear_quant_w(
    const float* __restrict__ w, __bf16* __restrict__ wq)
{
    const int group = blockIdx.x * 8 + (threadIdx.x >> 5);
    const int lane  = threadIdx.x & 31;

    const float4 v = reinterpret_cast<const float4*>(w)[(size_t)group * 32 + lane];
    float s = fabsf(v.x) + fabsf(v.y) + fabsf(v.z) + fabsf(v.w);
#pragma unroll
    for (int off = 16; off > 0; off >>= 1) s += __shfl_xor(s, off, 32);

    const float scale = fmaxf(s * (1.0f / 128.0f), 1e-8f);
    const float thr   = 0.5f * scale;
    const unsigned short bp = f2bf(scale);
    const unsigned short bn = bp | 0x8000u;

    auto pick = [&](float x) -> unsigned {
        return (x > thr) ? (unsigned)bp : ((x < -thr) ? (unsigned)bn : 0u);
    };
    uint2 o;
    o.x = pick(v.x) | (pick(v.y) << 16);
    o.y = pick(v.z) | (pick(v.w) << 16);
    reinterpret_cast<uint2*>(wq)[(size_t)group * 32 + lane] = o;
}

// -------------------- kernel 2: x f32 -> bf16 ------------------------------
__global__ __launch_bounds__(256) void bitlinear_cvt_x(
    const float4* __restrict__ x, uint2* __restrict__ xb, int n4)
{
    const int i = blockIdx.x * blockDim.x + threadIdx.x;
    if (i >= n4) return;
    const float4 v = x[i];
    uint2 o;
    o.x = (unsigned)f2bf(v.x) | ((unsigned)f2bf(v.y) << 16);
    o.y = (unsigned)f2bf(v.z) | ((unsigned)f2bf(v.w) << 16);
    xb[i] = o;
}

// -------------------- TDM: rows x 32-halves tile, global -> padded LDS -----
// D# per ISA ch.8: 2-D tile, data_size=2B, row = 64B data + 16B LDS pad.
__device__ __forceinline__ void tdm_load_tile(unsigned lds_byte_off,
                                              const __bf16* gptr, int rows) {
    const unsigned long long ga = (unsigned long long)(uintptr_t)gptr;

    u32x4 g0;
    g0.x = 1u;                                           // count=1 (valid), load
    g0.y = lds_byte_off;                                 // lds_addr
    g0.z = (unsigned)(ga & 0xffffffffu);                 // global_addr[31:0]
    g0.w = (unsigned)((ga >> 32) & 0x1ffffffu)           // global_addr[56:32]
         | (2u << 30);                                   // type = 2 ("image")

    i32x8 g1;
    g1[0] = (1 << 16)        // data_size = 2 bytes
          | (1 << 20)        // pad_enable
          | (3 << 22)        // pad_interval: 16 DWORDs (64 B) between pads
          | (3 << 25);       // pad_amount:   4 DWORDs (16 B)
    g1[1] = (GEMM_K & 0xffff) << 16;                     // tensor_dim0[15:0]
    g1[2] = (GEMM_K >> 16) | (rows << 16);               // dim0 hi | tensor_dim1
    g1[3] = (KSTEP << 16);                               // dim1 hi=0 | tile_dim0=32
    g1[4] = rows;                                        // tile_dim1, tile_dim2=0
    g1[5] = GEMM_K;                                      // tensor_dim0_stride lo
    g1[6] = 0;                                           // stride hi | dim1_stride lo
    g1[7] = 0;

    const i32x4 z4 = (i32x4)0;
#if defined(__clang_major__) && (__clang_major__ >= 23)
    const i32x8 z8 = (i32x8)0;
    __builtin_amdgcn_tensor_load_to_lds(g0, g1, z4, z4, z8, 0);
#else
    __builtin_amdgcn_tensor_load_to_lds(g0, g1, z4, z4, 0);
#endif
}

__device__ __forceinline__ unsigned lds_off(const void* p) {
    // low 32 bits of the LDS-aperture flat address == LDS byte offset
    return (unsigned)(uintptr_t)p;
}

// -------------------- fragment loaders (ISA 7.12.2 layouts) ----------------
__device__ __forceinline__ bf16x16 frag_a(const __bf16* base, int row, int lane) {
    const __bf16* p = base + row * LDSK + ((lane < 16) ? 0 : 8);
    FragU f;
    f.u[0] = *reinterpret_cast<const uint4*>(p);
    f.u[1] = *reinterpret_cast<const uint4*>(p + 16);
    return f.v;
}
__device__ __forceinline__ bf16x16 frag_b(const __bf16* base, int col, int lane) {
    const __bf16* p = base + col * LDSK + ((lane < 16) ? 0 : 8);
    FragU f;
    f.u[0] = *reinterpret_cast<const uint4*>(p);
    f.u[1] = *reinterpret_cast<const uint4*>(p + 16);
    return f.v;
}

// -------------------- kernel 3: WMMA GEMM + bias, TDM-staged ---------------
__global__ __launch_bounds__(256) void bitlinear_gemm(
    const __bf16* __restrict__ A,   // x_bf16, MxK row-major
    const __bf16* __restrict__ B,   // wq_bf16, NxK row-major
    const float*  __restrict__ bias,
    float*        __restrict__ C)   // MxN row-major
{
    __shared__ __align__(16) __bf16 As[2][TILE_M * LDSK];   // 2 x 20 KB
    __shared__ __align__(16) __bf16 Bs[2][TILE_N * LDSK];   // 2 x 10 KB

    const int n0   = blockIdx.x * TILE_N;
    const int m0   = blockIdx.y * TILE_M;
    const int tid  = threadIdx.x;
    const int lane = tid & 31;
    const int wave = tid >> 5;
    const int wm   = (wave >> 1) * 64;   // 0/64/128/192
    const int wn   = (wave & 1) * 64;    // 0/64

    const __bf16* Ab = A + (size_t)m0 * GEMM_K;
    const __bf16* Bb = B + (size_t)n0 * GEMM_K;

    const unsigned ldsA[2] = { lds_off(&As[0][0]), lds_off(&As[1][0]) };
    const unsigned ldsB[2] = { lds_off(&Bs[0][0]), lds_off(&Bs[1][0]) };

    f32x8 acc[4][4];
#pragma unroll
    for (int i = 0; i < 4; ++i)
#pragma unroll
        for (int j = 0; j < 4; ++j) acc[i][j] = (f32x8)(0.0f);

    // wave 0 drives the Tensor Data Mover (EXEC-independent, per-wave op)
    if (wave == 0) {
        tdm_load_tile(ldsA[0], Ab, TILE_M);
        tdm_load_tile(ldsB[0], Bb, TILE_N);
    }

    const int iters = GEMM_K / KSTEP;
    for (int it = 0; it < iters; ++it) {
        const int cur = it & 1;
        const int nxt = cur ^ 1;
        const int knext = (it + 1) * KSTEP;

        // buffer `nxt` was last read in iteration it-1: fence before overwrite
        __syncthreads();
        if (wave == 0) {
            if (knext < GEMM_K) {
                tdm_load_tile(ldsA[nxt], Ab + knext, TILE_M);
                tdm_load_tile(ldsB[nxt], Bb + knext, TILE_N);
                __builtin_amdgcn_s_wait_tensorcnt(2);  // current pair landed
            } else {
                __builtin_amdgcn_s_wait_tensorcnt(0);
            }
        }
        __syncthreads();   // publish LDS tile to all 8 waves

        const __bf16* aB = &As[cur][0];
        const __bf16* bB = &Bs[cur][0];

        bf16x16 af[4], bfm[4];
#pragma unroll
        for (int i = 0; i < 4; ++i) af[i]  = frag_a(aB, wm + i * 16 + (lane & 15), lane);
#pragma unroll
        for (int j = 0; j < 4; ++j) bfm[j] = frag_b(bB, wn + j * 16 + (lane & 15), lane);

#pragma unroll
        for (int i = 0; i < 4; ++i)
#pragma unroll
            for (int j = 0; j < 4; ++j)
                acc[i][j] = __builtin_amdgcn_wmma_f32_16x16x32_bf16(
                    false, af[i], false, bfm[j], (short)0, acc[i][j], false, false);
    }

    // Epilogue: C layout: VGPR r, lanes 0-15 -> M=r, lanes 16-31 -> M=8+r; N=lane%16
#pragma unroll
    for (int j = 0; j < 4; ++j) {
        const int col = n0 + wn + j * 16 + (lane & 15);
        const float bv = bias[col];
#pragma unroll
        for (int i = 0; i < 4; ++i) {
            const int rbase = m0 + wm + i * 16 + ((lane < 16) ? 0 : 8);
#pragma unroll
            for (int r = 0; r < 8; ++r)
                C[(size_t)(rbase + r) * GEMM_N + col] = acc[i][j][r] + bv;
        }
    }
}

// ---------------------------------------------------------------------------
extern "C" void kernel_launch(void* const* d_in, const int* in_sizes, int n_in,
                              void* d_out, int out_size, void* d_ws, size_t ws_size,
                              hipStream_t stream) {
    const float* x    = (const float*)d_in[0];   // (4,2048,4096) f32
    const float* w    = (const float*)d_in[1];   // (16384,4096)  f32
    const float* bias = (const float*)d_in[2];   // (16384,)      f32
    float* out = (float*)d_out;                  // (4,2048,16384) f32

    __bf16* xb = (__bf16*)d_ws;
    __bf16* wb = (__bf16*)((char*)d_ws + (size_t)GEMM_M * GEMM_K * sizeof(__bf16));

    const int n4 = GEMM_M * GEMM_K / 4;
    bitlinear_cvt_x<<<n4 / 256, 256, 0, stream>>>(
        (const float4*)x, (uint2*)xb, n4);

    const int ngroups = GEMM_N * (GEMM_K / 128);
    bitlinear_quant_w<<<ngroups / 8, 256, 0, stream>>>(w, wb);

    dim3 grid(GEMM_N / TILE_N, GEMM_M / TILE_M);   // (128, 32)
    bitlinear_gemm<<<grid, 256, 0, stream>>>(xb, wb, bias, out);
}
